// BFeatVanillaGAT_43215960932950
// MI455X (gfx1250) — matrix operand order repaired and checked by reading, not compiled, not run on hardware
//
#include <hip/hip_runtime.h>
#include <stdint.h>

// ---------------------------------------------------------------------------
// BFeatVanillaGAT forward for MI455X (gfx1250).
// Compute-bound (~370 GFLOP vs ~1GB traffic) -> all large GEMMs run on
// v_wmma_f32_16x16x32_bf16 (f32 accumulate). Gathers/concats are fused into
// the GEMM A-operand; small MLPs / softmax / layernorm run on VALU with
// LDS-staged weights.
// ---------------------------------------------------------------------------

#define NNODES 1024
#define NEDGES 64512
#define DMODEL 512
#define NHEADS 8

typedef __bf16 bf16_t;
typedef __attribute__((ext_vector_type(16))) __bf16 v16bf;
typedef __attribute__((ext_vector_type(8)))  __bf16 v8bf;
typedef __attribute__((ext_vector_type(8)))  float   v8f;

// ---- bf16 <-> f32 helpers (round-to-nearest-even), bit-exact, no __bf16 math
__device__ __forceinline__ unsigned short f2bfbits(float f) {
  unsigned u = __float_as_uint(f);
  u += 0x7FFFu + ((u >> 16) & 1u);
  return (unsigned short)(u >> 16);
}
__device__ __forceinline__ bf16_t f2bf(float f) {
  unsigned short s = f2bfbits(f);
  bf16_t b;
  __builtin_memcpy(&b, &s, 2);
  return b;
}
__device__ __forceinline__ float bfbits2f(unsigned short s) {
  return __uint_as_float(((unsigned)s) << 16);
}
__device__ __forceinline__ float bf2f(bf16_t b) {
  unsigned short s;
  __builtin_memcpy(&s, &b, 2);
  return bfbits2f(s);
}

// ---------------------------------------------------------------------------
// Generic elementwise converts / fills
// ---------------------------------------------------------------------------
__global__ void cvt_f32_to_bf16(const float* __restrict__ in,
                                bf16_t* __restrict__ out, long long n) {
  long long i = (long long)blockIdx.x * blockDim.x + threadIdx.x;
  if (i < n) out[i] = f2bf(in[i]);
}

// W is (K x N) f32 row-major; write Wt = (N x K) bf16 row-major
__global__ void wt_transpose_bf16(const float* __restrict__ W,
                                  bf16_t* __restrict__ Wt, int K, int N) {
  long long i = (long long)blockIdx.x * blockDim.x + threadIdx.x;
  long long total = (long long)K * N;
  if (i < total) {
    int k = (int)(i / N), n = (int)(i % N);
    Wt[(size_t)n * K + k] = f2bf(W[i]);
  }
}

__global__ void fill_u32(unsigned* __restrict__ p, unsigned v, long long n) {
  long long i = (long long)blockIdx.x * blockDim.x + threadIdx.x;
  if (i < n) p[i] = v;
}

// ---------------------------------------------------------------------------
// Tiled bf16 WMMA GEMM.  C[M,N] = act(A[M,K] @ Wt[N,K]^T + bias)
// Wave computes a 16x32 tile (two accumulators); workgroup (8 waves) = 64x64.
// A-operand modes:
//   0: direct rows of A0 (row stride K)
//   1: rows A0[idx0[r]]            (row stride 512)   -- x[dst] for pv
//   2: concat [A0[idx0[r]] | A1[r] | A0[idx1[r]]]     -- [x_i | ef | x_j]
//   3: concat [A0[r] | A1[r]] (each 512 wide)         -- [x | agg]
// M, N multiples of 64; K multiple of 32 (holds for all calls here).
// ---------------------------------------------------------------------------
template <int AMODE, bool RELU, bool OUTF, bool OUTB>
__global__ __launch_bounds__(256) void gemm_wmma_bf16(
    const bf16_t* __restrict__ A0, const bf16_t* __restrict__ A1,
    const int* __restrict__ idx0, const int* __restrict__ idx1,
    const bf16_t* __restrict__ Wt, const float* __restrict__ bias,
    float* __restrict__ outF, bf16_t* __restrict__ outB,
    int M, int N, int K) {
  const int t    = threadIdx.x & 31;
  const int wid  = threadIdx.x >> 5;
  const int r16  = t & 15;
  const int half = t >> 4;                 // wave32 half select
  const int tileM = blockIdx.y * 64 + (wid & 3) * 16;
  const int tileN = blockIdx.x * 64 + (wid >> 2) * 32;
  const int aRow  = tileM + r16;

  int gi0 = 0, gi1 = 0;
  if (AMODE == 1 || AMODE == 2) gi0 = idx0[aRow];
  if (AMODE == 2)               gi1 = idx1[aRow];

  const int col0 = tileN + r16;
  const int col1 = tileN + 16 + r16;
  const bf16_t* bp0base = Wt + (size_t)col0 * K;
  const bf16_t* bp1base = Wt + (size_t)col1 * K;

  v8f acc0 = {};
  v8f acc1 = {};

  for (int kb = 0; kb < K; kb += 32) {
    const bf16_t* ap;
    if (AMODE == 0) {
      ap = A0 + (size_t)aRow * K + kb;
    } else if (AMODE == 1) {
      ap = A0 + (size_t)gi0 * 512 + kb;
    } else if (AMODE == 2) {
      if (kb < 512)       ap = A0 + (size_t)gi0 * 512 + kb;
      else if (kb < 1024) ap = A1 + (size_t)aRow * 512 + (kb - 512);
      else                ap = A0 + (size_t)gi1 * 512 + (kb - 1024);
    } else {  // AMODE == 3
      if (kb < 512) ap = A0 + (size_t)aRow * 512 + kb;
      else          ap = A1 + (size_t)aRow * 512 + (kb - 512);
    }
    // A fragment (16x32 bf16): half 0 lanes carry K 0..15,
    // half 1 lanes carry K 16..31 (matches B)
    v16bf af = *(const v16bf*)(ap + half * 16);
    // B fragments (32x16 bf16): lane = column, halves carry K 0..15 / 16..31.
    // Wt is N x K so these are contiguous 32B loads.
    v16bf bf0 = *(const v16bf*)(bp0base + kb + half * 16);
    v16bf bf1 = *(const v16bf*)(bp1base + kb + half * 16);
    acc0 = __builtin_amdgcn_wmma_f32_16x16x32_bf16(false, af, false, bf0,
                                                   (short)0, acc0, false, false);
    acc1 = __builtin_amdgcn_wmma_f32_16x16x32_bf16(false, af, false, bf1,
                                                   (short)0, acc1, false, false);
  }

  const float b0 = bias ? bias[col0] : 0.f;
  const float b1 = bias ? bias[col1] : 0.f;
#pragma unroll
  for (int j = 0; j < 8; ++j) {
    const int rowO = tileM + j + (half ? 8 : 0);   // C/D layout (ISA 7.12.2)
    float v0 = acc0[j] + b0;
    float v1 = acc1[j] + b1;
    if (RELU) { v0 = fmaxf(v0, 0.f); v1 = fmaxf(v1, 0.f); }
    const size_t o0 = (size_t)rowO * N + col0;
    const size_t o1 = (size_t)rowO * N + col1;
    if (OUTF) { outF[o0] = v0; outF[o1] = v1; }
    if (OUTB) { outB[o0] = f2bf(v0); outB[o1] = f2bf(v1); }
  }
}

// ---------------------------------------------------------------------------
// Pairwise distance-bias MLP: 4 -> 32 (relu,LN) -> 32 (relu,LN) -> 8
// One thread per (a,b) pair; weights staged in LDS.
// ---------------------------------------------------------------------------
__device__ __forceinline__ void ln32(float* h, const float* g, const float* be) {
  float m = 0.f;
#pragma unroll
  for (int i = 0; i < 32; ++i) m += h[i];
  m *= (1.f / 32.f);
  float v = 0.f;
#pragma unroll
  for (int i = 0; i < 32; ++i) { float d = h[i] - m; v += d * d; }
  v *= (1.f / 32.f);
  const float r = rsqrtf(v + 1e-5f);
#pragma unroll
  for (int i = 0; i < 32; ++i) h[i] = (h[i] - m) * r * g[i] + be[i];
}

__global__ __launch_bounds__(256) void dist_mlp_kernel(
    const float* __restrict__ center, const int* __restrict__ batch_ids,
    const float* __restrict__ f1w, const float* __restrict__ f1b,
    const float* __restrict__ f1g, const float* __restrict__ f1be,
    const float* __restrict__ f2w, const float* __restrict__ f2b,
    const float* __restrict__ f2g, const float* __restrict__ f2be,
    const float* __restrict__ f3w, const float* __restrict__ f3b,
    float* __restrict__ dw) {
  __shared__ float s1w[4 * 32], s2w[32 * 32], s3w[32 * 8];
  __shared__ float s1b[32], s1g[32], s1be[32], s2b[32], s2g[32], s2be[32], s3b[8];
  const int tid = threadIdx.x;
  for (int i = tid; i < 128; i += 256)  s1w[i] = f1w[i];
  for (int i = tid; i < 1024; i += 256) s2w[i] = f2w[i];
  for (int i = tid; i < 256; i += 256)  s3w[i] = f3w[i];
  if (tid < 32) {
    s1b[tid] = f1b[tid]; s1g[tid] = f1g[tid]; s1be[tid] = f1be[tid];
    s2b[tid] = f2b[tid]; s2g[tid] = f2g[tid]; s2be[tid] = f2be[tid];
  }
  if (tid < 8) s3b[tid] = f3b[tid];
  __syncthreads();

  const int id = blockIdx.x * 256 + tid;           // < 1024*1024
  const int a = id >> 10, b = id & 1023;
  const float dx = center[b * 3 + 0] - center[a * 3 + 0];
  const float dy = center[b * 3 + 1] - center[a * 3 + 1];
  const float dz = center[b * 3 + 2] - center[a * 3 + 2];
  const float dist = sqrtf(dx * dx + dy * dy + dz * dz);
  const float w[4] = {dx, dy, dz, dist};

  float h[32];
#pragma unroll 4
  for (int o = 0; o < 32; ++o) {
    float s = s1b[o];
#pragma unroll
    for (int i = 0; i < 4; ++i) s += w[i] * s1w[i * 32 + o];
    h[o] = fmaxf(s, 0.f);
  }
  ln32(h, s1g, s1be);
  float h2[32];
#pragma unroll 4
  for (int o = 0; o < 32; ++o) {
    float s = s2b[o];
    for (int c = 0; c < 32; ++c) s += h[c] * s2w[c * 32 + o];
    h2[o] = fmaxf(s, 0.f);
  }
  ln32(h2, s2g, s2be);

  const bool same = batch_ids[a] == batch_ids[b];
#pragma unroll
  for (int hh = 0; hh < 8; ++hh) {
    float s = s3b[hh];
    for (int c = 0; c < 32; ++c) s += h2[c] * s3w[c * 8 + hh];
    dw[(size_t)id * 8 + hh] = same ? s : 0.f;
  }
}

// ---------------------------------------------------------------------------
// Masked multi-head self-attention core: per (head, graph) block of 64 rows.
// Mask == same-graph block, which is exactly this 64x64 tile.
// ---------------------------------------------------------------------------
__global__ __launch_bounds__(64) void attn_kernel(
    const float* __restrict__ qf, const float* __restrict__ kf,
    const float* __restrict__ vf, const float* __restrict__ dw,
    bf16_t* __restrict__ attnbf) {
  const int h = blockIdx.x;     // head
  const int g = blockIdx.y;     // graph
  const int n = threadIdx.x;    // row within graph
  __shared__ float ks[64][64];
  __shared__ float vs[64][64];
  const int base = g * 64;
  for (int d = 0; d < 64; ++d) {
    ks[n][d] = kf[(size_t)(base + n) * 512 + h * 64 + d];
    vs[n][d] = vf[(size_t)(base + n) * 512 + h * 64 + d];
  }
  __syncthreads();
  float qr[64];
  for (int d = 0; d < 64; ++d) qr[d] = qf[(size_t)(base + n) * 512 + h * 64 + d];
  float sc[64];
  float mx = -1e30f;
  for (int m = 0; m < 64; ++m) {
    float s = 0.f;
    for (int d = 0; d < 64; ++d) s += qr[d] * ks[m][d];
    s = s * 0.125f + dw[((size_t)(base + n) * 1024 + (base + m)) * 8 + h];
    sc[m] = s;
    mx = fmaxf(mx, s);
  }
  float sum = 0.f;
  for (int m = 0; m < 64; ++m) { sc[m] = __expf(sc[m] - mx); sum += sc[m]; }
  const float inv = 1.f / sum;
  for (int d = 0; d < 64; ++d) {
    float o = 0.f;
    for (int m = 0; m < 64; ++m) o += sc[m] * vs[m][d];
    attnbf[(size_t)(base + n) * 512 + h * 64 + d] = f2bf(o * inv);
  }
}

// ---------------------------------------------------------------------------
// x = LayerNorm(obj + oproj) * g + b  -> bf16.  One block per row.
// ---------------------------------------------------------------------------
__global__ __launch_bounds__(256) void residual_ln_kernel(
    const float* __restrict__ obj, const float* __restrict__ opr,
    const float* __restrict__ g, const float* __restrict__ b,
    bf16_t* __restrict__ xbf) {
  __shared__ float red[256];
  __shared__ float stat[2];
  const int tid = threadIdx.x;
  const int row = blockIdx.x;
  const int c0 = tid * 2, c1 = tid * 2 + 1;
  const float a0 = obj[(size_t)row * 512 + c0] + opr[(size_t)row * 512 + c0];
  const float a1 = obj[(size_t)row * 512 + c1] + opr[(size_t)row * 512 + c1];
  red[tid] = a0 + a1;
  __syncthreads();
  for (int s = 128; s > 0; s >>= 1) {
    if (tid < s) red[tid] += red[tid + s];
    __syncthreads();
  }
  if (tid == 0) stat[0] = red[0] * (1.f / 512.f);
  __syncthreads();
  const float mean = stat[0];
  const float d0 = a0 - mean, d1 = a1 - mean;
  red[tid] = d0 * d0 + d1 * d1;
  __syncthreads();
  for (int s = 128; s > 0; s >>= 1) {
    if (tid < s) red[tid] += red[tid + s];
    __syncthreads();
  }
  if (tid == 0) stat[1] = rsqrtf(red[0] * (1.f / 512.f) + 1e-5f);
  __syncthreads();
  const float r = stat[1];
  xbf[(size_t)row * 512 + c0] = f2bf(d0 * r * g[c0] + b[c0]);
  xbf[(size_t)row * 512 + c1] = f2bf(d1 * r * g[c1] + b[c1]);
}

// ---------------------------------------------------------------------------
// Fused per-edge head attention: a=[qh|eh] -> relu(c1@a) -> c2 -> softmax(o)
// -> msg = prob*val -> segment max (order-preserving uint atomicMax).
// Persistent blocks; c1/c2 weights staged once per block in LDS (bf16 bits).
// ---------------------------------------------------------------------------
__device__ __forceinline__ unsigned ordEncode(float f) {
  unsigned u = __float_as_uint(f);
  return (u & 0x80000000u) ? ~u : (u | 0x80000000u);
}

__global__ __launch_bounds__(256) void edge_attn_kernel(
    const bf16_t* __restrict__ pqx, const bf16_t* __restrict__ pex,
    const bf16_t* __restrict__ val, const float* __restrict__ c1w,
    const float* __restrict__ c1b, const float* __restrict__ c2w,
    const float* __restrict__ c2b, const int* __restrict__ src,
    unsigned* __restrict__ aggu) {
  __shared__ unsigned short sc1w[128 * 128];   // 32 KB
  __shared__ unsigned short sc2w[64 * 128];    // 16 KB
  __shared__ float sc1b[128], sc2b[64];
  __shared__ float sa[1024], sh1[1024], sh2[512];
  __shared__ float smax[8], sinv[8];
  const int tid = threadIdx.x;
  for (int i = tid; i < 128 * 128; i += 256) sc1w[i] = f2bfbits(c1w[i]);
  for (int i = tid; i < 64 * 128;  i += 256) sc2w[i] = f2bfbits(c2w[i]);
  if (tid < 128) sc1b[tid] = c1b[tid];
  if (tid < 64)  sc2b[tid] = c2b[tid];
  __syncthreads();

  for (int e = blockIdx.x; e < NEDGES; e += gridDim.x) {
    const int s = src[e];
    for (int i = tid; i < 512; i += 256) {
      sa[i]       = bf2f(pqx[(size_t)s * 512 + i]);   // qh: (c,h)=row[c*8+h]
      sa[512 + i] = bf2f(pex[(size_t)e * 512 + i]);   // eh
    }
    __syncthreads();
    for (int i = tid; i < 1024; i += 256) {          // c1: 128 out x 8 heads
      const int o = i >> 3, hh = i & 7;
      float acc = sc1b[o];
      const unsigned short* wr = &sc1w[o * 128];
      for (int c = 0; c < 128; ++c) acc += bfbits2f(wr[c]) * sa[c * 8 + hh];
      sh1[i] = fmaxf(acc, 0.f);
    }
    __syncthreads();
    for (int i = tid; i < 512; i += 256) {           // c2: 64 out x 8 heads
      const int o = i >> 3, hh = i & 7;
      float acc = sc2b[o];
      const unsigned short* wr = &sc2w[o * 128];
      for (int c = 0; c < 128; ++c) acc += bfbits2f(wr[c]) * sh1[c * 8 + hh];
      sh2[i] = acc;
    }
    __syncthreads();
    if (tid < 8) {                                   // softmax over o, per head
      float m = -1e30f;
      for (int o = 0; o < 64; ++o) m = fmaxf(m, sh2[o * 8 + tid]);
      float su = 0.f;
      for (int o = 0; o < 64; ++o) su += __expf(sh2[o * 8 + tid] - m);
      smax[tid] = m;
      sinv[tid] = 1.f / su;
    }
    __syncthreads();
    for (int j = tid; j < 512; j += 256) {
      const int hh = j & 7;
      const float p = __expf(sh2[j] - smax[hh]) * sinv[hh];
      const float m = p * bf2f(val[(size_t)e * 512 + j]);
      atomicMax(&aggu[(size_t)s * 512 + j], ordEncode(m));
    }
    __syncthreads();
  }
}

__global__ void agg_decode_kernel(const unsigned* __restrict__ aggu,
                                  bf16_t* __restrict__ aggbf, long long n) {
  long long i = (long long)blockIdx.x * blockDim.x + threadIdx.x;
  if (i >= n) return;
  const unsigned u = aggu[i];
  float f;
  if (u == 0u) f = 0.f;                        // never touched / -inf -> 0
  else if (u & 0x80000000u) f = __uint_as_float(u ^ 0x80000000u);
  else f = __uint_as_float(~u);
  if (f < -1e29f) f = f;                       // (keep negatives; no clamp)
  aggbf[i] = f2bf(f);
}

// ---------------------------------------------------------------------------
// Host orchestration
// ---------------------------------------------------------------------------
enum {
  IN_OBJ = 0, IN_EDGEF, IN_CENTER, IN_EIDX, IN_BATCH,
  P_C1_B, P_C1_W, P_C2_B, P_C2_W,
  P_E1_B, P_E1_W, P_E2_B, P_E2_W,
  P_F1_B, P_F1_BETA, P_F1_G, P_F1_W,
  P_F2_B, P_F2_BETA, P_F2_G, P_F2_W,
  P_F3_B, P_F3_W,
  P_K_B, P_K_W, P_LN_B, P_LN_G, P_O_B, P_O_W,
  P_P1_B, P_P1_W, P_P2_B, P_P2_W,
  P_PE_B, P_PE_W, P_PQ_B, P_PQ_W, P_PV_B, P_PV_W,
  P_Q_B, P_Q_W, P_V_B, P_V_W
};

static inline int blks(long long n, int b) { return (int)((n + b - 1) / b); }

extern "C" void kernel_launch(void* const* d_in, const int* in_sizes, int n_in,
                              void* d_out, int out_size, void* d_ws,
                              size_t ws_size, hipStream_t stream) {
  (void)in_sizes; (void)n_in; (void)out_size; (void)ws_size;

  const float* obj    = (const float*)d_in[IN_OBJ];
  const float* edgef  = (const float*)d_in[IN_EDGEF];
  const float* center = (const float*)d_in[IN_CENTER];
  const int*   eidx   = (const int*)d_in[IN_EIDX];
  const int*   batch  = (const int*)d_in[IN_BATCH];
  const int*   src    = eidx;
  const int*   dst    = eidx + NEDGES;

  // ---- workspace carve-out
  char* ws = (char*)d_ws;
  size_t off = 0;
  auto take = [&](size_t bytes) -> char* {
    char* p = ws + off;
    off += (bytes + 255) & ~(size_t)255;
    return p;
  };
  bf16_t* wq_t  = (bf16_t*)take((size_t)512 * 512 * 2);
  bf16_t* wk_t  = (bf16_t*)take((size_t)512 * 512 * 2);
  bf16_t* wv_t  = (bf16_t*)take((size_t)512 * 512 * 2);
  bf16_t* wo_t  = (bf16_t*)take((size_t)512 * 512 * 2);
  bf16_t* we1_t = (bf16_t*)take((size_t)1024 * 1536 * 2);
  bf16_t* we2_t = (bf16_t*)take((size_t)512 * 1024 * 2);
  bf16_t* wpq_t = (bf16_t*)take((size_t)512 * 512 * 2);
  bf16_t* wpe_t = (bf16_t*)take((size_t)512 * 512 * 2);
  bf16_t* wpv_t = (bf16_t*)take((size_t)512 * 512 * 2);
  bf16_t* wp1_t = (bf16_t*)take((size_t)1024 * 1024 * 2);
  bf16_t* wp2_t = (bf16_t*)take((size_t)512 * 1024 * 2);
  bf16_t* objbf = (bf16_t*)take((size_t)NNODES * 512 * 2);
  bf16_t* ebf   = (bf16_t*)take((size_t)NEDGES * 512 * 2);
  float*  dwb   = (float*)take((size_t)NNODES * NNODES * 8 * 4);
  float*  qf    = (float*)take((size_t)NNODES * 512 * 4);
  float*  kf    = (float*)take((size_t)NNODES * 512 * 4);
  float*  vf    = (float*)take((size_t)NNODES * 512 * 4);
  bf16_t* attnb = (bf16_t*)take((size_t)NNODES * 512 * 2);
  float*  oproj = (float*)take((size_t)NNODES * 512 * 4);
  bf16_t* xbf   = (bf16_t*)take((size_t)NNODES * 512 * 2);
  bf16_t* ef1bf = (bf16_t*)take((size_t)NEDGES * 1024 * 2);
  bf16_t* valbf = (bf16_t*)take((size_t)NEDGES * 512 * 2);
  bf16_t* pqxbf = (bf16_t*)take((size_t)NNODES * 512 * 2);
  bf16_t* pexbf = (bf16_t*)take((size_t)NEDGES * 512 * 2);
  unsigned* aggu = (unsigned*)take((size_t)NNODES * 512 * 4);
  bf16_t* aggbf = (bf16_t*)take((size_t)NNODES * 512 * 2);
  bf16_t* h1bf  = (bf16_t*)take((size_t)NNODES * 1024 * 2);

  float* out_xo = (float*)d_out;                       // (1024, 512)
  float* out_ef = (float*)d_out + (size_t)NNODES * 512; // (64512, 512)

  // ---- 0) init aggregation buffer
  fill_u32<<<blks((long long)NNODES * 512, 256), 256, 0, stream>>>(
      aggu, 0u, (long long)NNODES * 512);

  // ---- 1) weight transposes -> bf16 (N x K)
  auto wt = [&](int pidx, bf16_t* dstp, int K, int N) {
    wt_transpose_bf16<<<blks((long long)K * N, 256), 256, 0, stream>>>(
        (const float*)d_in[pidx], dstp, K, N);
  };
  wt(P_Q_W, wq_t, 512, 512);   wt(P_K_W, wk_t, 512, 512);
  wt(P_V_W, wv_t, 512, 512);   wt(P_O_W, wo_t, 512, 512);
  wt(P_E1_W, we1_t, 1536, 1024); wt(P_E2_W, we2_t, 1024, 512);
  wt(P_PQ_W, wpq_t, 512, 512); wt(P_PE_W, wpe_t, 512, 512);
  wt(P_PV_W, wpv_t, 512, 512); wt(P_P1_W, wp1_t, 1024, 1024);
  wt(P_P2_W, wp2_t, 1024, 512);

  // ---- 2) activation converts
  cvt_f32_to_bf16<<<blks((long long)NNODES * 512, 256), 256, 0, stream>>>(
      obj, objbf, (long long)NNODES * 512);
  cvt_f32_to_bf16<<<blks((long long)NEDGES * 512, 256), 256, 0, stream>>>(
      edgef, ebf, (long long)NEDGES * 512);

  // ---- 3) pairwise distance-bias MLP
  dist_mlp_kernel<<<(NNODES * NNODES) / 256, 256, 0, stream>>>(
      center, batch,
      (const float*)d_in[P_F1_W], (const float*)d_in[P_F1_B],
      (const float*)d_in[P_F1_G], (const float*)d_in[P_F1_BETA],
      (const float*)d_in[P_F2_W], (const float*)d_in[P_F2_B],
      (const float*)d_in[P_F2_G], (const float*)d_in[P_F2_BETA],
      (const float*)d_in[P_F3_W], (const float*)d_in[P_F3_B], dwb);

  // ---- 4) QKV projections (WMMA)
  dim3 gNode(512 / 64, NNODES / 64);
  gemm_wmma_bf16<0, false, true, false><<<gNode, 256, 0, stream>>>(
      objbf, nullptr, nullptr, nullptr, wq_t, (const float*)d_in[P_Q_B],
      qf, nullptr, NNODES, 512, 512);
  gemm_wmma_bf16<0, false, true, false><<<gNode, 256, 0, stream>>>(
      objbf, nullptr, nullptr, nullptr, wk_t, (const float*)d_in[P_K_B],
      kf, nullptr, NNODES, 512, 512);
  gemm_wmma_bf16<0, false, true, false><<<gNode, 256, 0, stream>>>(
      objbf, nullptr, nullptr, nullptr, wv_t, (const float*)d_in[P_V_B],
      vf, nullptr, NNODES, 512, 512);

  // ---- 5) attention core + O projection + residual LayerNorm
  attn_kernel<<<dim3(NHEADS, 16), 64, 0, stream>>>(qf, kf, vf, dwb, attnb);
  gemm_wmma_bf16<0, false, true, false><<<gNode, 256, 0, stream>>>(
      attnb, nullptr, nullptr, nullptr, wo_t, (const float*)d_in[P_O_B],
      oproj, nullptr, NNODES, 512, 512);
  residual_ln_kernel<<<NNODES, 256, 0, stream>>>(
      obj, oproj, (const float*)d_in[P_LN_G], (const float*)d_in[P_LN_B], xbf);

  // ---- 6) edge feature MLP: ef = relu([x_i|ef|x_j] @ e1) @ e2 ; relu -> out
  gemm_wmma_bf16<2, true, false, true><<<dim3(1024 / 64, NEDGES / 64), 256, 0,
                                         stream>>>(
      xbf, ebf, src, dst, we1_t, (const float*)d_in[P_E1_B],
      nullptr, ef1bf, NEDGES, 1024, 1536);
  gemm_wmma_bf16<0, true, true, false><<<dim3(512 / 64, NEDGES / 64), 256, 0,
                                         stream>>>(
      ef1bf, nullptr, nullptr, nullptr, we2_t, (const float*)d_in[P_E2_B],
      out_ef, nullptr, NEDGES, 512, 1024);

  // ---- 7) per-edge attention inputs: val = x_j@pv, qh = x@pq, eh = ef@pe
  gemm_wmma_bf16<1, false, false, true><<<dim3(512 / 64, NEDGES / 64), 256, 0,
                                          stream>>>(
      xbf, nullptr, dst, nullptr, wpv_t, (const float*)d_in[P_PV_B],
      nullptr, valbf, NEDGES, 512, 512);
  gemm_wmma_bf16<0, false, false, true><<<gNode, 256, 0, stream>>>(
      xbf, nullptr, nullptr, nullptr, wpq_t, (const float*)d_in[P_PQ_B],
      nullptr, pqxbf, NNODES, 512, 512);
  gemm_wmma_bf16<0, false, false, true><<<dim3(512 / 64, NEDGES / 64), 256, 0,
                                          stream>>>(
      ebf, nullptr, nullptr, nullptr, wpe_t, (const float*)d_in[P_PE_B],
      nullptr, pexbf, NEDGES, 512, 512);

  // ---- 8) fused c1/c2/softmax/message + segment-max aggregation
  edge_attn_kernel<<<1024, 256, 0, stream>>>(
      pqxbf, pexbf, valbf, (const float*)d_in[P_C1_W],
      (const float*)d_in[P_C1_B], (const float*)d_in[P_C2_W],
      (const float*)d_in[P_C2_B], src, aggu);
  agg_decode_kernel<<<blks((long long)NNODES * 512, 256), 256, 0, stream>>>(
      aggu, aggbf, (long long)NNODES * 512);

  // ---- 9) output MLP: xo = relu(relu([x|agg] @ p1) @ p2) -> d_out
  gemm_wmma_bf16<3, true, false, true><<<dim3(1024 / 64, NNODES / 64), 256, 0,
                                         stream>>>(
      xbf, aggbf, nullptr, nullptr, wp1_t, (const float*)d_in[P_P1_B],
      nullptr, h1bf, NNODES, 1024, 1024);
  gemm_wmma_bf16<0, true, true, false><<<gNode, 256, 0, stream>>>(
      h1bf, nullptr, nullptr, nullptr, wp2_t, (const float*)d_in[P_P2_B],
      out_xo, nullptr, NNODES, 512, 1024);
}
